// SVDLinear_16896401342477
// MI455X (gfx1250) — matrix operand order repaired and checked by reading, not compiled, not run on hardware
//
#include <hip/hip_runtime.h>
#include <hip/hip_bf16.h>

typedef __bf16 v4bf  __attribute__((ext_vector_type(4)));
typedef __bf16 v8bf  __attribute__((ext_vector_type(8)));
typedef __bf16 v16bf __attribute__((ext_vector_type(16)));
typedef float  v8f   __attribute__((ext_vector_type(8)));
typedef int    v4i   __attribute__((ext_vector_type(4)));

#define D_DIM 4096
#define RANK  819
#define RP    832          // rank padded to multiple of 64
#define M_TOK 16384        // B*T = 8*2048

#define BM 128
#define BK 32
#define LDS_STRIDE 40      // halves; 32 + 8 pad -> 80B row stride, conflict-free b128 reads

#if defined(__has_builtin)
#if __has_builtin(__builtin_amdgcn_global_load_async_to_lds_b128)
#define HAVE_ASYNC_LDS 1
#endif
#endif

__device__ __forceinline__ void async_copy16(const __bf16* g, __bf16* l) {
#ifdef HAVE_ASYNC_LDS
    __builtin_amdgcn_global_load_async_to_lds_b128((v4i*)g, (v4i*)l, 0, 0);
#else
    *(uint4*)l = *(const uint4*)g;
#endif
}

__device__ __forceinline__ void async_wait() {
#ifdef HAVE_ASYNC_LDS
#if __has_builtin(__builtin_amdgcn_s_wait_asynccnt)
    __builtin_amdgcn_s_wait_asynccnt(0);
#else
    asm volatile("s_wait_asynccnt 0x0" ::: "memory");
#endif
#endif
}

__device__ __forceinline__ v16bf frag_combine(v8bf lo, v8bf hi) {
    return __builtin_shufflevector(lo, hi, 0,1,2,3,4,5,6,7,8,9,10,11,12,13,14,15);
}

// UsT[r][d] = bf16(U[d][r] * S[r]), zero rows for r >= RANK (pad)
__global__ void prep_usT(const float* __restrict__ U, const float* __restrict__ S,
                         __bf16* __restrict__ UsT) {
    int idx = blockIdx.x * blockDim.x + threadIdx.x;   // over RP*D
    int r = idx / D_DIM;
    int d = idx - r * D_DIM;
    float v = 0.0f;
    if (r < RANK) v = U[(size_t)d * RANK + r] * S[r];
    UsT[idx] = (__bf16)v;
}

// VT[d][r] = bf16(V[r][d]), zero for r >= RANK (pad K of GEMM2)
__global__ void prep_vT(const float* __restrict__ V, __bf16* __restrict__ VT) {
    int idx = blockIdx.x * blockDim.x + threadIdx.x;   // over D*RP
    int d = idx / RP;
    int r = idx - d * RP;
    float v = (r < RANK) ? V[(size_t)r * D_DIM + d] : 0.0f;
    VT[idx] = (__bf16)v;
}

// C[M,N] = A[M,K] @ BT[N,K]^T  (BT stored row-major N x K, K contiguous)
// A_F32: A is f32 (converted to bf16 on the fly), else A is bf16 (async to LDS).
// EPI:   write f32 C = acc + alpha * Xres; else write bf16 C = acc.
template<int BN_T, bool A_F32, bool EPI>
__global__ __launch_bounds__(256)
void wmma_gemm(const void* __restrict__ Aptr, const __bf16* __restrict__ BT,
               void* __restrict__ Cptr, const float* __restrict__ Xres,
               const float* __restrict__ alphaPtr, int K, int N)
{
    constexpr int NSUB = BN_T / 32;   // 16-wide n-subtiles per wave (2 waves across N)

    __shared__ __bf16 As[2][BM * LDS_STRIDE];
    __shared__ __bf16 Bs[2][BN_T * LDS_STRIDE];

    const int tid  = threadIdx.x;
    const int lane = tid & 31;
    const int wave = tid >> 5;
    const int wm   = wave & 3;        // 32-row slice within 128
    const int wn   = wave >> 2;       // half of BN_T
    const int m0   = blockIdx.x * BM;
    const int n0   = blockIdx.y * BN_T;

    const int khalf = lane >> 4;      // 0 or 1
    const int l15   = lane & 15;

    const float*  Af = (const float*)Aptr;
    const __bf16* Ab = (const __bf16*)Aptr;

    v8f acc[2][NSUB] = {};
    float4 areg[4];

    // ---- staging helpers ----
    auto load_A_f32 = [&](int kk) {           // issue global loads only
        #pragma unroll
        for (int it = 0; it < 4; ++it) {
            int slot = tid + it * 256;        // 0..1023
            int row  = slot >> 3;
            int c4   = slot & 7;
            areg[it] = *(const float4*)&Af[(size_t)(m0 + row) * K + kk + c4 * 4];
        }
    };
    auto commit_A_f32 = [&](int buf) {        // convert + ds_store
        #pragma unroll
        for (int it = 0; it < 4; ++it) {
            int slot = tid + it * 256;
            int row  = slot >> 3;
            int c4   = slot & 7;
            v4bf b;
            b[0] = (__bf16)areg[it].x; b[1] = (__bf16)areg[it].y;
            b[2] = (__bf16)areg[it].z; b[3] = (__bf16)areg[it].w;
            *(v4bf*)&As[buf][row * LDS_STRIDE + c4 * 4] = b;
        }
    };
    auto stage_A_async = [&](int kk, int buf) {
        #pragma unroll
        for (int it = 0; it < 2; ++it) {
            int slot = tid + it * 256;        // 0..511
            int row  = slot >> 2;
            int c    = slot & 3;
            async_copy16(&Ab[(size_t)(m0 + row) * K + kk + c * 8],
                         &As[buf][row * LDS_STRIDE + c * 8]);
        }
    };
    auto stage_B_async = [&](int kk, int buf) {
        #pragma unroll
        for (int it = 0; it < (BN_T * 4) / 256; ++it) {
            int slot = tid + it * 256;
            int row  = slot >> 2;
            int c    = slot & 3;
            async_copy16(&BT[(size_t)(n0 + row) * K + kk + c * 8],
                         &Bs[buf][row * LDS_STRIDE + c * 8]);
        }
    };
    auto do_math = [&](int buf) {
        v16bf afrag[2], bfrag[NSUB];
        #pragma unroll
        for (int mi = 0; mi < 2; ++mi) {
            const __bf16* base = &As[buf][(wm * 32 + mi * 16 + l15) * LDS_STRIDE];
            afrag[mi] = frag_combine(*(const v8bf*)&base[khalf * 8],
                                     *(const v8bf*)&base[16 + khalf * 8]);
        }
        #pragma unroll
        for (int ni = 0; ni < NSUB; ++ni) {
            const __bf16* base = &Bs[buf][(wn * (BN_T / 2) + ni * 16 + l15) * LDS_STRIDE];
            bfrag[ni] = frag_combine(*(const v8bf*)&base[khalf * 8],
                                     *(const v8bf*)&base[16 + khalf * 8]);
        }
        #pragma unroll
        for (int mi = 0; mi < 2; ++mi)
            #pragma unroll
            for (int ni = 0; ni < NSUB; ++ni)
                acc[mi][ni] = __builtin_amdgcn_wmma_f32_16x16x32_bf16(
                    false, afrag[mi], false, bfrag[ni],
                    (short)0, acc[mi][ni], false, false);
    };

    // ---- prologue: stage tile 0 into buffer 0 ----
    if (A_F32) { load_A_f32(0); commit_A_f32(0); }
    else       { stage_A_async(0, 0); }
    stage_B_async(0, 0);
    async_wait();
    __syncthreads();

    // ---- pipelined main loop ----
    const int nk = K / BK;
    for (int k = 0; k < nk; ++k) {
        const int cur = k & 1;
        const int nxt = cur ^ 1;
        const bool has_next = (k + 1 < nk);
        if (has_next) {
            if (A_F32) load_A_f32((k + 1) * BK);
            else       stage_A_async((k + 1) * BK, nxt);
            stage_B_async((k + 1) * BK, nxt);
        }
        do_math(cur);
        if (has_next && A_F32) commit_A_f32(nxt);
        if (has_next) async_wait();
        __syncthreads();
    }

    // ---- epilogue ----
    float alpha = 0.0f;
    if (EPI) alpha = *alphaPtr;

    #pragma unroll
    for (int mi = 0; mi < 2; ++mi) {
        #pragma unroll
        for (int ni = 0; ni < NSUB; ++ni) {
            const int mbase = m0 + wm * 32 + mi * 16 + 8 * khalf;
            const int nn    = n0 + wn * (BN_T / 2) + ni * 16 + l15;
            #pragma unroll
            for (int i = 0; i < 8; ++i) {
                const int m = mbase + i;
                float v = acc[mi][ni][i];
                if (EPI) {
                    float xr = Xres[(size_t)m * N + nn];
                    ((float*)Cptr)[(size_t)m * N + nn] = v + alpha * xr;
                } else {
                    ((__bf16*)Cptr)[(size_t)m * N + nn] = (__bf16)v;
                }
            }
        }
    }
}

extern "C" void kernel_launch(void* const* d_in, const int* in_sizes, int n_in,
                              void* d_out, int out_size, void* d_ws, size_t ws_size,
                              hipStream_t stream) {
    (void)in_sizes; (void)n_in; (void)out_size; (void)ws_size;

    const float* x     = (const float*)d_in[0];   // (8, 2048, 4096)
    const float* U     = (const float*)d_in[1];   // (4096, 819)
    const float* S     = (const float*)d_in[2];   // (819,)
    const float* V     = (const float*)d_in[3];   // (819, 4096)
    const float* alpha = (const float*)d_in[4];   // scalar
    float*       out   = (float*)d_out;           // (8, 2048, 4096) f32

    // workspace layout (bf16): UsT [RP x D], VT [D x RP], Y [M x RP]
    char* ws = (char*)d_ws;
    __bf16* UsT = (__bf16*)ws;
    __bf16* VT  = (__bf16*)(ws + (size_t)RP * D_DIM * 2);
    __bf16* Y   = (__bf16*)(ws + (size_t)RP * D_DIM * 4);

    const int prep_elems = RP * D_DIM;            // 832*4096
    prep_usT<<<prep_elems / 256, 256, 0, stream>>>(U, S, UsT);
    prep_vT <<<prep_elems / 256, 256, 0, stream>>>(V, VT);

    // GEMM1: Y[M, RP] = x[M, D] @ UsT^T   (K = D, BN = 64 since RP = 13*64)
    dim3 g1(M_TOK / BM, RP / 64);
    wmma_gemm<64, true, false><<<g1, dim3(256), 0, stream>>>(
        (const void*)x, UsT, (void*)Y, nullptr, nullptr, D_DIM, RP);

    // GEMM2: out[M, D] = Y[M, RP] @ VT^T + alpha * x   (K = RP, BN = 128)
    dim3 g2(M_TOK / BM, D_DIM / 128);
    wmma_gemm<128, false, true><<<g2, dim3(256), 0, stream>>>(
        (const void*)Y, VT, (void*)out, x, alpha, RP, D_DIM);
}
